// Model_81423989997722
// MI455X (gfx1250) — matrix operand order repaired and checked by reading, not compiled, not run on hardware
//
#include <hip/hip_runtime.h>

// ---------------------------------------------------------------------------
// Types for CDNA5 WMMA
// ---------------------------------------------------------------------------
typedef __attribute__((ext_vector_type(16))) _Float16 v16h;
typedef __attribute__((ext_vector_type(8)))  float    v8f;

union Frag16 {
    v16h         v;
    _Float16     h[16];
    unsigned int u[8];
};

// ---------------------------------------------------------------------------
// Problem constants
// ---------------------------------------------------------------------------
#define NB      4          // batch
#define BANDS   128
#define TIME    128
#define PER     8
#define PB      130        // padded bands
#define PT      130        // padded time
#define PP      10         // padded per
#define PSTRIDE (PB*PT*PP) // 169000 per batch
#define NPOS    (BANDS*TIME*PER)   // 131072 positions per batch
#define NATOM   512
#define K27     27
#define K32     32
#define NITERS  128
#define NS      32768      // output samples

// ---------------------------------------------------------------------------
// Zero residual + sparse buffers
// ---------------------------------------------------------------------------
__global__ void k_zero2(float* __restrict__ a, float* __restrict__ b, int n) {
    int gid = blockIdx.x * 256 + threadIdx.x;
    if (gid < n) { a[gid] = 0.0f; b[gid] = 0.0f; }
}

// ---------------------------------------------------------------------------
// h = x @ embed_w + embed_b   written into interior of padded residual
// ---------------------------------------------------------------------------
__global__ void k_embed(const float* __restrict__ x, const float* __restrict__ ew,
                        const float* __restrict__ eb, float* __restrict__ residual) {
    int gid = blockIdx.x * 256 + threadIdx.x;              // 4*128*128 rows
    if (gid >= NB * BANDS * TIME) return;
    int b    = gid >> 14;
    int rem  = gid & 16383;
    int band = rem >> 7;
    int t    = rem & 127;
    const float* xr = x + (size_t)gid * 257;
    float acc[8];
#pragma unroll
    for (int j = 0; j < 8; ++j) acc[j] = eb[j];
    for (int k = 0; k < 257; ++k) {
        float xv = xr[k];
#pragma unroll
        for (int j = 0; j < 8; ++j) acc[j] += xv * ew[k * 8 + j];
    }
    float* rr = residual + b * PSTRIDE + band * (PT * PP) + t * PP;
#pragma unroll
    for (int j = 0; j < 8; ++j) rr[j] = acc[j];
}

// ---------------------------------------------------------------------------
// Pack atoms (512x27 f32) -> (512x32 f16), zero pad K 27..31
// ---------------------------------------------------------------------------
__global__ void k_atoms_pack(const float* __restrict__ atoms, _Float16* __restrict__ ah) {
    int gid = blockIdx.x * 256 + threadIdx.x;   // 512*32
    if (gid >= NATOM * K32) return;
    int a = gid >> 5, k = gid & 31;
    ah[gid] = (k < K27) ? (_Float16)atoms[a * K27 + k] : (_Float16)0.0f;
}

// ---------------------------------------------------------------------------
// residual f32 -> f16  (runs ONCE; incremental updates patch resh in place)
// ---------------------------------------------------------------------------
__global__ void k_r2h(const float* __restrict__ r, _Float16* __restrict__ h, int n) {
    int gid = blockIdx.x * 256 + threadIdx.x;
    if (gid < n) h[gid] = (_Float16)r[gid];
}

__global__ void k_init_best(unsigned long long* best) {
    if (threadIdx.x < NB) best[threadIdx.x] = 0ull;
}

// ---------------------------------------------------------------------------
// Core: conv(residual, atoms) as WMMA GEMM fused with global argmax.
//   D[a,p] = sum_k atoms_f16[a,k] * patch_f16[k,p]
// Block = 256 threads = 8 waves. Each wave owns TWO 16-position N-tiles
// (A fragment reused by 2 WMMAs) and loops all 32 atom M-tiles.
// In-loop argmax is a v_max tree per tile tracking (value, tile) only;
// the row within the winning tile is recovered in k_sparse_update.
// ---------------------------------------------------------------------------
__global__ __launch_bounds__(256)
void k_sparse_step(const _Float16* __restrict__ resh,
                   const _Float16* __restrict__ atomsh,
                   unsigned long long* __restrict__ best) {
    __shared__ _Float16 sA[NATOM * K32];   // 32768 B
    const int tid = threadIdx.x;
    {   // cooperative 128b-wide load of all atoms into LDS
        const uint4* src = (const uint4*)atomsh;
        uint4*       dst = (uint4*)sA;
        for (int i = tid; i < NATOM * K32 / 8; i += 256) dst[i] = src[i];
    }
    __syncthreads();

    const int b       = blockIdx.y;
    const int wave    = tid >> 5;
    const int lane    = tid & 31;
    const int n       = lane & 15;
    const int hi_half = lane >> 4;
    const int kbase   = hi_half * 8;                  // 0 or 8
    const int p0      = (blockIdx.x * 8 + wave) * 32; // gridDim.x = 512
    const int pA      = p0 + n;
    const int pB      = p0 + 16 + n;

    // Branchless B-fragment builder (clamped address + cndmask zero).
    const _Float16* rbase = resh + b * PSTRIDE;
    Frag16 fbA, fbB;
    {
        int bandA = pA >> 10, remA = pA & 1023, tA = remA >> 3, perA = remA & 7;
        int bandB = pB >> 10, remB = pB & 1023, tB = remB >> 3, perB = remB & 7;
        const _Float16* rbA = rbase + bandA * (PT * PP) + tA * PP + perA;
        const _Float16* rbB = rbase + bandB * (PT * PP) + tB * PP + perB;
#pragma unroll
        for (int i = 0; i < 16; ++i) {
            int k  = kbase + (i & 7) + ((i & 8) << 1);
            int kc = (k < K27) ? k : 0;
            int dz = kc / 9, r9 = kc - dz * 9;
            int dy = r9 / 3, dx = r9 - dy * 3;
            int off = dz * (PT * PP) + dy * PP + dx;
            _Float16 vA = rbA[off];
            _Float16 vB = rbB[off];
            fbA.h[i] = (k < K27) ? vA : (_Float16)0.0f;
            fbB.h[i] = (k < K27) ? vB : (_Float16)0.0f;
        }
    }

    float bmaxA = -3.4e38f, bmaxB = -3.4e38f;
    int   bmtA  = 0,        bmtB  = 0;
    const unsigned int* sa32 = (const unsigned int*)sA;

    for (int mt = 0; mt < 32; ++mt) {
        // A fragment: 16x32 atom tile, row = mt*16 + (lane&15).
        // lanes 0-15: K {0..7,16..23}; lanes 16-31: K {8..15,24..31}.
        int row    = mt * 16 + (lane & 15);
        int dwbase = row * 16 + (hi_half ? 4 : 0);
        Frag16 fa;
#pragma unroll
        for (int i = 0; i < 4; ++i) {
            fa.u[i]     = sa32[dwbase + i];
            fa.u[4 + i] = sa32[dwbase + 8 + i];
        }
        v8f cA = {};
        cA = __builtin_amdgcn_wmma_f32_16x16x32_f16(
                 false, fa.v, false, fbA.v, (short)0, cA, false, false);
        v8f cB = {};
        cB = __builtin_amdgcn_wmma_f32_16x16x32_f16(
                 false, fa.v, false, fbB.v, (short)0, cB, false, false);

        float tA = fmaxf(fmaxf(fmaxf(cA[0], cA[1]), fmaxf(cA[2], cA[3])),
                         fmaxf(fmaxf(cA[4], cA[5]), fmaxf(cA[6], cA[7])));
        float tB = fmaxf(fmaxf(fmaxf(cB[0], cB[1]), fmaxf(cB[2], cB[3])),
                         fmaxf(fmaxf(cB[4], cB[5]), fmaxf(cB[6], cB[7])));
        if (tA > bmaxA) { bmaxA = tA; bmtA = mt; }
        if (tB > bmaxB) { bmaxB = tB; bmtB = mt; }
    }

    // Combine the two N-tiles, then order-preserving (value, ~idx) encode.
    float        bmax = bmaxA;
    unsigned int bidx = ((unsigned)bmtA << 17) | (unsigned)pA;
    if (bmaxB > bmax) {
        bmax = bmaxB;
        bidx = ((unsigned)bmtB << 17) | (unsigned)pB;
    }
    unsigned int u     = __float_as_uint(bmax);
    unsigned int key32 = (u & 0x80000000u) ? ~u : (u | 0x80000000u);
    unsigned long long key =
        ((unsigned long long)key32 << 32) | (unsigned long long)(~bidx);
    atomicMax(&best[b], key);
}

// ---------------------------------------------------------------------------
// Resolve the winning row within the best (tile, column), then apply:
// residual -= val*atom, sparse += val*atom, patch resh, reset best.
// One wave per batch (block = 128).
// ---------------------------------------------------------------------------
__global__ void k_sparse_update(unsigned long long* __restrict__ best,
                                const float* __restrict__ atoms,
                                float* __restrict__ residual,
                                float* __restrict__ sparse,
                                _Float16* __restrict__ resh) {
    __shared__ float dots[NB][16];
    __shared__ int   win[NB];
    int b    = threadIdx.x >> 5;
    int lane = threadIdx.x & 31;
    unsigned long long key = best[b];
    unsigned int idx = ~(unsigned int)(key & 0xFFFFFFFFu);
    unsigned int mt  = (idx >> 17) & 31u;
    unsigned int p   = idx & (NPOS - 1);
    int band = p >> 10;
    int rem  = p & 1023;
    int t    = rem >> 3;
    int per  = rem & 7;
    const float* rb = residual + b * PSTRIDE + band * (PT * PP) + t * PP + per;

    // Recompute the 16 candidate dot products of the winning tile/column.
    if (lane < 16) {
        int atom = mt * 16 + lane;
        float s = 0.0f;
#pragma unroll
        for (int k = 0; k < K27; ++k) {
            int dz = k / 9, r9 = k - dz * 9;
            int dy = r9 / 3, dx = r9 - dy * 3;
            s += atoms[atom * K27 + k] * rb[dz * (PT * PP) + dy * PP + dx];
        }
        dots[b][lane] = s;
    }
    __syncthreads();
    if (lane == 0) {
        int   br = 0;
        float bv = dots[b][0];
#pragma unroll
        for (int r = 1; r < 16; ++r)
            if (dots[b][r] > bv) { bv = dots[b][r]; br = r; }
        win[b]  = br;
        best[b] = 0ull;            // reset for next iteration
    }
    __syncthreads();
    int   atom = mt * 16 + win[b];
    float val  = dots[b][win[b]];
    if (lane < K27) {
        int dz = lane / 9, r9 = lane - dz * 9;
        int dy = r9 / 3,  dx = r9 - dy * 3;
        float av  = val * atoms[atom * K27 + lane];
        int   off = b * PSTRIDE + (band + dz) * (PT * PP) + (t + dy) * PP + (per + dx);
        float nv  = residual[off] - av;
        residual[off] = nv;
        resh[off]     = (_Float16)nv;   // keep f16 mirror in sync
        sparse[off]  += av;
    }
}

// ---------------------------------------------------------------------------
// enc[b, per*128+band, t] = sparse[b, band, t, per]
// ---------------------------------------------------------------------------
__global__ void k_enc(const float* __restrict__ sparse, float* __restrict__ enc) {
    int gid = blockIdx.x * 256 + threadIdx.x;          // 4*1024*128
    if (gid >= NB * 1024 * TIME) return;
    int b   = gid >> 17;
    int rem = gid & 131071;
    int ch  = rem >> 7;
    int t   = rem & 127;
    int per  = ch >> 7;
    int band = ch & 127;
    enc[gid] = sparse[b * PSTRIDE + band * (PT * PP) + t * PP + per];
}

__global__ void k_ctx(const float* __restrict__ enc, float* __restrict__ ctx) {
    int gid = blockIdx.x * 256 + threadIdx.x;          // 4096
    if (gid >= NB * 1024) return;
    const float* e = enc + (size_t)gid * TIME;
    float s = 0.0f;
    for (int t = 0; t < TIME; ++t) s += e[t];
    ctx[gid] = s;
}

// ---------------------------------------------------------------------------
// Generic FC: out[b,j] = bias[j] + sum_k in[b,k] * w[k,j]
// ---------------------------------------------------------------------------
__global__ void k_fc(const float* __restrict__ in, const float* __restrict__ w,
                     const float* __restrict__ bias, float* __restrict__ out,
                     int K, int N, int total) {
    int gid = blockIdx.x * 256 + threadIdx.x;
    if (gid >= total) return;
    int b = gid / N, j = gid - b * N;
    float acc = bias[j];
    const float* ib = in + (size_t)b * K;
    for (int k = 0; k < K; ++k) acc += ib[k] * w[(size_t)k * N + j];
    out[gid] = acc;
}

// ---------------------------------------------------------------------------
// LayerNorm over last axis + leaky relu (leak=1.0 -> identity activation).
// grid = rows, block = 256.
// ---------------------------------------------------------------------------
__global__ void k_ln_act(const float* __restrict__ in, float* __restrict__ out,
                         const float* __restrict__ g, const float* __restrict__ be,
                         int N, float leak) {
    int row = blockIdx.x;
    const float* x = in + (size_t)row * N;
    __shared__ float red[256];
    float s = 0.0f;
    for (int j = threadIdx.x; j < N; j += 256) s += x[j];
    red[threadIdx.x] = s; __syncthreads();
    for (int st = 128; st > 0; st >>= 1) {
        if (threadIdx.x < st) red[threadIdx.x] += red[threadIdx.x + st];
        __syncthreads();
    }
    float mu = red[0] / N;
    __syncthreads();
    float s2 = 0.0f;
    for (int j = threadIdx.x; j < N; j += 256) { float d = x[j] - mu; s2 += d * d; }
    red[threadIdx.x] = s2; __syncthreads();
    for (int st = 128; st > 0; st >>= 1) {
        if (threadIdx.x < st) red[threadIdx.x] += red[threadIdx.x + st];
        __syncthreads();
    }
    float inv = rsqrtf(red[0] / N + 1e-5f);
    for (int j = threadIdx.x; j < N; j += 256) {
        float v = (x[j] - mu) * inv * g[j] + be[j];
        out[(size_t)row * N + j] = v > 0.0f ? v : leak * v;
    }
}

// ---------------------------------------------------------------------------
// conv1d, kernel 7, same padding. grid = (ceil(Co*T/256), B)
// ---------------------------------------------------------------------------
__global__ void k_conv1d(const float* __restrict__ in, const float* __restrict__ w,
                         const float* __restrict__ bias, float* __restrict__ out,
                         int Ci, int Co, int T) {
    int gid = blockIdx.x * 256 + threadIdx.x;
    int b   = blockIdx.y;
    if (gid >= Co * T) return;
    int co = gid / T, t = gid - co * T;
    float acc = bias[co];
    const float* ib = in + (size_t)b * Ci * T;
    const float* wb = w + (size_t)co * Ci * 7;
    for (int ci = 0; ci < Ci; ++ci) {
        const float* ir = ib + (size_t)ci * T;
        const float* wr = wb + ci * 7;
#pragma unroll
        for (int dk = 0; dk < 7; ++dk) {
            int tt = t + dk - 3;
            if (tt >= 0 && tt < T) acc += ir[tt] * wr[dk];
        }
    }
    out[((size_t)b * Co + co) * T + t] = acc;
}

// ---------------------------------------------------------------------------
// Per-channel batchnorm stats over (batch, time). grid = Co blocks.
// ---------------------------------------------------------------------------
__global__ void k_bn_stats(const float* __restrict__ x, float* __restrict__ stats,
                           int Co, int T, int B) {
    int co = blockIdx.x;
    __shared__ float r1[256], r2[256];
    float s = 0.0f, s2 = 0.0f;
    for (int i = threadIdx.x; i < B * T; i += 256) {
        int b = i / T, t = i - b * T;
        float v = x[((size_t)b * Co + co) * T + t];
        s += v; s2 += v * v;
    }
    r1[threadIdx.x] = s; r2[threadIdx.x] = s2; __syncthreads();
    for (int st = 128; st > 0; st >>= 1) {
        if (threadIdx.x < st) {
            r1[threadIdx.x] += r1[threadIdx.x + st];
            r2[threadIdx.x] += r2[threadIdx.x + st];
        }
        __syncthreads();
    }
    if (threadIdx.x == 0) {
        float mu  = r1[0] / (B * T);
        float var = r2[0] / (B * T) - mu * mu;
        stats[co * 2]     = mu;
        stats[co * 2 + 1] = var;
    }
}

// ---------------------------------------------------------------------------
// batchnorm + leaky + repeat4 upsample
// ---------------------------------------------------------------------------
__global__ void k_bn_up4(const float* __restrict__ x, const float* __restrict__ stats,
                         const float* __restrict__ g, const float* __restrict__ be,
                         float* __restrict__ y, int Co, int T) {
    int gid = blockIdx.x * 256 + threadIdx.x;
    int b   = blockIdx.y;
    if (gid >= Co * T) return;
    int co = gid / T, t = gid - co * T;
    float mu = stats[co * 2], var = stats[co * 2 + 1];
    float v = (x[((size_t)b * Co + co) * T + t] - mu) * rsqrtf(var + 1e-5f) * g[co] + be[co];
    v = v > 0.0f ? v : 0.2f * v;
    float* yo = y + ((size_t)b * Co + co) * (size_t)(T * 4) + (size_t)t * 4;
    yo[0] = v; yo[1] = v; yo[2] = v; yo[3] = v;
}

// ---------------------------------------------------------------------------
// softmax over 2 logits per batch
// ---------------------------------------------------------------------------
__global__ void k_softmax2(const float* __restrict__ raw, float* __restrict__ mix) {
    int b = threadIdx.x;
    if (b < NB) {
        float a = raw[b * 2], c = raw[b * 2 + 1];
        float m = fmaxf(a, c);
        float ea = __expf(a - m), ec = __expf(c - m);
        float inv = 1.0f / (ea + ec);
        mix[b * 2] = ea * inv; mix[b * 2 + 1] = ec * inv;
    }
}

// ---------------------------------------------------------------------------
// wet = causal convolution of dry with imp (== rfft-conv first NS samples)
// grid = (NS/256, B), block = 256, LDS-tiled.
// ---------------------------------------------------------------------------
__global__ __launch_bounds__(256)
void k_wet(const float* __restrict__ dry, const float* __restrict__ imp,
           float* __restrict__ wet) {
    const int b   = blockIdx.y;
    const int tid = threadIdx.x;
    const int n0  = blockIdx.x * 256;
    const int n   = n0 + tid;
    __shared__ float sd[256];
    __shared__ float si[512];
    float acc = 0.0f;
    for (int c0 = 0; c0 <= n0; c0 += 256) {
        sd[tid] = dry[b * NS + c0 + tid];
        int base = n0 - c0 - 255;
        for (int j = tid; j < 511; j += 256) {
            int m = base + j;
            si[j] = (m >= 0 && m < NS) ? imp[b * NS + m] : 0.0f;
        }
        __syncthreads();
        for (int j = 0; j < 256; ++j) {
            int k = c0 + j;
            if (k <= n) acc += sd[j] * si[tid + 255 - j];
        }
        __syncthreads();
    }
    wet[b * NS + n] = acc;
}

__global__ void k_mixout(const float* __restrict__ dry, const float* __restrict__ wet,
                         const float* __restrict__ mix, float* __restrict__ out) {
    int gid = blockIdx.x * 256 + threadIdx.x;     // 4*32768
    if (gid >= NB * NS) return;
    int b = gid >> 15;
    out[gid] = mix[b * 2] * dry[gid] + mix[b * 2 + 1] * wet[gid];
}

// ---------------------------------------------------------------------------
// Host orchestration
// ---------------------------------------------------------------------------
static inline size_t alignup(size_t x) { return (x + 255) & ~(size_t)255; }

extern "C" void kernel_launch(void* const* d_in, const int* in_sizes, int n_in,
                              void* d_out, int out_size, void* d_ws, size_t ws_size,
                              hipStream_t stream) {
    (void)in_sizes; (void)n_in; (void)out_size; (void)ws_size;

    const float* x         = (const float*)d_in[0];
    const float* atoms     = (const float*)d_in[2];
    const float* embed_w   = (const float*)d_in[3];
    const float* embed_b   = (const float*)d_in[4];
    const float* ctx_w[3]  = { (const float*)d_in[5],  (const float*)d_in[9],  (const float*)d_in[13] };
    const float* ctx_b[3]  = { (const float*)d_in[6],  (const float*)d_in[10], (const float*)d_in[14] };
    const float* ctx_g[3]  = { (const float*)d_in[7],  (const float*)d_in[11], (const float*)d_in[15] };
    const float* ctx_be[3] = { (const float*)d_in[8],  (const float*)d_in[12], (const float*)d_in[16] };
    const float* up_w[5]   = { (const float*)d_in[17], (const float*)d_in[21], (const float*)d_in[25],
                               (const float*)d_in[29], (const float*)d_in[33] };
    const float* up_b[5]   = { (const float*)d_in[18], (const float*)d_in[22], (const float*)d_in[26],
                               (const float*)d_in[30], (const float*)d_in[34] };
    const float* bn_g[4]   = { (const float*)d_in[19], (const float*)d_in[23], (const float*)d_in[27],
                               (const float*)d_in[31] };
    const float* bn_b[4]   = { (const float*)d_in[20], (const float*)d_in[24], (const float*)d_in[28],
                               (const float*)d_in[32] };
    const float* verb_ln_g = (const float*)d_in[35];
    const float* verb_ln_b = (const float*)d_in[36];
    const float* verb_imp_w = (const float*)d_in[37];
    const float* verb_imp_b = (const float*)d_in[38];
    const float* verb_mix_w = (const float*)d_in[39];
    const float* verb_mix_b = (const float*)d_in[40];

    // ---- workspace carve-out ----
    char* ws = (char*)d_ws;
    size_t off = 0;
    float* residual = (float*)(ws + off); off = alignup(off + (size_t)NB * PSTRIDE * 4);
    float* sparse   = (float*)(ws + off); off = alignup(off + (size_t)NB * PSTRIDE * 4);
    _Float16* resh  = (_Float16*)(ws + off); off = alignup(off + (size_t)NB * PSTRIDE * 2);
    _Float16* atomsh = (_Float16*)(ws + off); off = alignup(off + (size_t)NATOM * K32 * 2);
    unsigned long long* best = (unsigned long long*)(ws + off); off = alignup(off + NB * 8);
    float* enc   = (float*)(ws + off); off = alignup(off + (size_t)NB * 1024 * TIME * 4);
    float* ctx   = (float*)(ws + off); off = alignup(off + NB * 1024 * 4);
    float* cbuf0 = (float*)(ws + off); off = alignup(off + NB * 1024 * 4);
    float* cbuf1 = (float*)(ws + off); off = alignup(off + NB * 1024 * 4);
    float* cc    = (float*)(ws + off); off = alignup(off + NB * 1024 * 4);
    float* stats = (float*)(ws + off); off = alignup(off + 512 * 2 * 4);
    float* dry   = (float*)(ws + off); off = alignup(off + (size_t)NB * NS * 4);
    float* imp   = (float*)(ws + off); off = alignup(off + (size_t)NB * NS * 4);
    float* wet   = (float*)(ws + off); off = alignup(off + (size_t)NB * NS * 4);
    float* mix   = (float*)(ws + off); off = alignup(off + NB * 2 * 4);
    float* yA    = (float*)(ws + off); off = alignup(off + (size_t)NB * 64 * 8192 * 4);   // max conv out
    float* yB    = (float*)(ws + off); off = alignup(off + (size_t)NB * 64 * NS * 4);     // max upsampled

    const int npad = NB * PSTRIDE;

    // ---- embed + pad ----
    k_zero2<<<(npad + 255) / 256, 256, 0, stream>>>(residual, sparse, npad);
    k_embed<<<(NB * BANDS * TIME + 255) / 256, 256, 0, stream>>>(x, embed_w, embed_b, residual);
    k_atoms_pack<<<(NATOM * K32 + 255) / 256, 256, 0, stream>>>(atoms, atomsh);
    k_r2h<<<(npad + 255) / 256, 256, 0, stream>>>(residual, resh, npad);   // once
    k_init_best<<<1, 32, 0, stream>>>(best);                               // once

    // ---- sparse_code: 128 WMMA matching-pursuit iterations ----
    for (int it = 0; it < NITERS; ++it) {
        k_sparse_step<<<dim3(NPOS / 256, NB), 256, 0, stream>>>(resh, atomsh, best);
        k_sparse_update<<<1, 128, 0, stream>>>(best, atoms, residual, sparse, resh);
    }

    // ---- enc + ctx ----
    k_enc<<<(NB * 1024 * TIME + 255) / 256, 256, 0, stream>>>(sparse, enc);
    k_ctx<<<(NB * 1024 + 255) / 256, 256, 0, stream>>>(enc, ctx);

    // ---- context MLP ----
    k_fc<<<16, 256, 0, stream>>>(ctx, ctx_w[0], ctx_b[0], cbuf0, 1024, 1024, NB * 1024);
    k_ln_act<<<NB, 256, 0, stream>>>(cbuf0, cbuf0, ctx_g[0], ctx_be[0], 1024, 0.2f);
    k_fc<<<16, 256, 0, stream>>>(cbuf0, ctx_w[1], ctx_b[1], cbuf1, 1024, 1024, NB * 1024);
    k_ln_act<<<NB, 256, 0, stream>>>(cbuf1, cbuf1, ctx_g[1], ctx_be[1], 1024, 0.2f);
    k_fc<<<16, 256, 0, stream>>>(cbuf1, ctx_w[2], ctx_b[2], cbuf0, 1024, 1024, NB * 1024);

    // ---- upsampling conv stack ----
    const int CiA[5] = {1024, 512, 256, 128, 64};
    const int CoA[5] = { 512, 256, 128,  64,  1};
    const int TA[5]  = { 128, 512, 2048, 8192, NS};
    const float* cur = enc;
    for (int i = 0; i < 4; ++i) {
        int Ci = CiA[i], Co = CoA[i], T = TA[i];
        k_conv1d<<<dim3((Co * T + 255) / 256, NB), 256, 0, stream>>>(cur, up_w[i], up_b[i], yA, Ci, Co, T);
        k_bn_stats<<<Co, 256, 0, stream>>>(yA, stats, Co, T, NB);
        k_bn_up4<<<dim3((Co * T + 255) / 256, NB), 256, 0, stream>>>(yA, stats, bn_g[i], bn_b[i], yB, Co, T);
        cur = yB;
    }
    k_conv1d<<<dim3((NS + 255) / 256, NB), 256, 0, stream>>>(cur, up_w[4], up_b[4], dry, 64, 1, NS);

    // ---- reverb head ----
    k_ln_act<<<NB, 256, 0, stream>>>(cbuf0, cc, verb_ln_g, verb_ln_b, 1024, 1.0f);
    k_fc<<<(NB * NS + 255) / 256, 256, 0, stream>>>(cc, verb_imp_w, verb_imp_b, imp, 1024, NS, NB * NS);
    k_fc<<<1, 256, 0, stream>>>(cc, verb_mix_w, verb_mix_b, cbuf1, 1024, 2, NB * 2);
    k_softmax2<<<1, 32, 0, stream>>>(cbuf1, mix);
    k_wet<<<dim3(NS / 256, NB), 256, 0, stream>>>(dry, imp, wet);
    k_mixout<<<(NB * NS + 255) / 256, 256, 0, stream>>>(dry, wet, mix, (float*)d_out);
}
